// ArkClassifier_5334349381952
// MI455X (gfx1250) — compile-verified, matrix operand order checked by reading, not run on hardware
//
#include <hip/hip_runtime.h>
#include <hip/hip_bf16.h>

// ---------------------------------------------------------------------------
// Swin-window attention block for MI455X (gfx1250), wave32 + WMMA bf16.
//   C=384, H=12, hd=32, N=144 tokens/window, B_=1152 windows, M=165888 rows.
// Pipeline: [convert weights] -> [QKV gemm + scatter] -> [attention] -> [proj]
// Matmuls: v_wmma_f32_16x16x32_bf16. Pure-copy LDS staging uses the CDNA5
// async path (global_load_async_to_lds_b128 + s_wait_asynccnt).
// ---------------------------------------------------------------------------

#define CDIM       384
#define NHEADS     12
#define HDIM       32
#define NTOK       144          // tokens per window
#define NWIN       144          // windows per image
#define BATCH      8
#define BWIN       (BATCH * NWIN)        // 1152
#define MROWS      (BWIN * NTOK)         // 165888
#define QKVN       (3 * CDIM)            // 1152
#define QSCALE     0.17677669529663687f  // 1/sqrt(32)

typedef __attribute__((ext_vector_type(16))) __bf16 v16bf;
typedef __attribute__((ext_vector_type(8)))  float  v8f;

union FragBF {
    v16bf v;
    unsigned short s[16];
    unsigned int   u[8];
};

__device__ __forceinline__ unsigned short f32_to_bf16(float x) {
    unsigned int u = __float_as_uint(x);
    u += 0x7FFFu + ((u >> 16) & 1u);   // round-to-nearest-even
    return (unsigned short)(u >> 16);
}

// CDNA5 async copy: 16B global -> LDS, tracked on ASYNCcnt (no VGPR round
// trip). dst must be an LDS pointer; generic-pointer low 32 bits are the
// wave-relative LDS byte address (flat LDS aperture).
__device__ __forceinline__ void async_copy_b128(void* lds_dst, const void* gsrc) {
    const unsigned lds = (unsigned)(unsigned long long)lds_dst;
    asm volatile("global_load_async_to_lds_b128 %0, %1, off"
                 :: "v"(lds), "v"(gsrc) : "memory");
}
__device__ __forceinline__ void async_wait0() {
    asm volatile("s_wait_asynccnt 0x0" ::: "memory");
}

// ---------------------------------------------------------------------------
// Fragment loaders (CDNA5 ISA 7.12.2, wave32):
//  A 16x32 bf16: lanes 0-15 hold rows M=0..15 with K{0..7,16..23} in v0..v7,
//                lanes 16-31 same rows with K{8..15,24..31}.
//  B 32x16 bf16 is the mirror over columns. Both are served by the same
//  helper when the LDS tile keeps the 32 contraction values contiguous per
//  row/column (tile[outer][k]).
// ---------------------------------------------------------------------------
__device__ __forceinline__ v16bf load_frag_rowK(const unsigned short* tile,
                                                int outer0, int ld) {
    const int lane = threadIdx.x & 31;
    const int half = lane >> 4;                 // 0: K 0-7/16-23, 1: K 8-15/24-31
    const unsigned short* p = tile + (outer0 + (lane & 15)) * ld;
    FragBF f;
#pragma unroll
    for (int i = 0; i < 4; ++i)
        f.u[i] = *(const unsigned int*)(p + (i * 2 + half * 8));
#pragma unroll
    for (int i = 0; i < 4; ++i)
        f.u[4 + i] = *(const unsigned int*)(p + (16 + i * 2 + half * 8));
    return f.v;
}

// Same A layout but sourced from an fp32 LDS tile (softmax probabilities),
// converting to bf16 while building the fragment.
__device__ __forceinline__ v16bf load_frag_rowK_f32(const float* tile,
                                                    int outer0, int ld) {
    const int lane = threadIdx.x & 31;
    const int half = lane >> 4;
    const float* p = tile + (outer0 + (lane & 15)) * ld;
    FragBF f;
#pragma unroll
    for (int i = 0; i < 8; ++i) {
        const int k = (i < 4) ? (i * 2 + half * 8) : (16 + (i - 4) * 2 + half * 8);
        f.s[2 * i]     = f32_to_bf16(p[k]);
        f.s[2 * i + 1] = f32_to_bf16(p[k + 1]);
    }
    return f.v;
}

// ---------------------------------------------------------------------------
// Kernel 0: fp32 -> bf16 weight conversion (qkv_w: 1152x384, proj_w: 384x384)
// ---------------------------------------------------------------------------
__global__ __launch_bounds__(256) void convert_weights_kernel(
    const float* __restrict__ qkv_w, const float* __restrict__ proj_w,
    unsigned short* __restrict__ qkvWb, unsigned short* __restrict__ projWb) {
    const int i = blockIdx.x * 256 + threadIdx.x;
    if (i < QKVN * CDIM) qkvWb[i] = f32_to_bf16(qkv_w[i]);
    if (i < CDIM * CDIM) projWb[i] = f32_to_bf16(proj_w[i]);
}

// ---------------------------------------------------------------------------
// Kernel 1: QKV GEMM  (M=165888, K=384, N=1152), epilogue scatters into
// head-major bf16 q/k/v buffers: [(b_*12+h)][token][hd]; Q pre-scaled.
// Tiling: 128x64 per WG, BK=32, 8 waves each owning a 32x32 patch.
// A (fp32 x) converted to bf16 while staging; B staged with async-LDS copy.
// ---------------------------------------------------------------------------
__global__ __launch_bounds__(256) void qkv_gemm_kernel(
    const float* __restrict__ X,             // (M, 384) fp32
    const unsigned short* __restrict__ Wb,   // (1152, 384) bf16, B[k][n]=Wb[n][k]
    const float* __restrict__ bias,          // (1152)
    unsigned short* __restrict__ q_ws,
    unsigned short* __restrict__ k_ws,
    unsigned short* __restrict__ v_ws) {
    __shared__ __align__(16) unsigned short sA[128 * 32];
    __shared__ __align__(16) unsigned short sB[64 * 32];

    const int tid  = threadIdx.x;
    const int wave = tid >> 5;
    const int m0   = blockIdx.y * 128;
    const int n0   = blockIdx.x * 64;
    const int wm   = (wave >> 1) * 32;   // wave row offset in tile
    const int wn   = (wave & 1) * 32;    // wave col offset in tile

    v8f acc[2][2] = {{{}, {}}, {{}, {}}};

    for (int kt = 0; kt < CDIM; kt += 32) {
        // Stage B: 64x32 bf16, pure copy -> async to LDS (ASYNCcnt).
        {
            const int n  = tid >> 2;
            const int kb = (tid & 3) * 8;
            async_copy_b128(&sB[n * 32 + kb], Wb + (n0 + n) * CDIM + kt + kb);
        }
        // Stage A: 128x32 fp32 -> bf16. Each thread: 16 contiguous elems.
        {
            const int row = tid >> 1;
            const int cb  = (tid & 1) * 16;
            const float* src = X + (m0 + row) * CDIM + kt + cb;
            if (kt + 32 < CDIM) __builtin_prefetch(src + 32, 0, 1);
#pragma unroll
            for (int j = 0; j < 4; ++j) {
                const float4 f = *(const float4*)(src + 4 * j);
                unsigned short* d = &sA[row * 32 + cb + 4 * j];
                d[0] = f32_to_bf16(f.x); d[1] = f32_to_bf16(f.y);
                d[2] = f32_to_bf16(f.z); d[3] = f32_to_bf16(f.w);
            }
        }
        async_wait0();
        __syncthreads();

        const v16bf a0 = load_frag_rowK(sA, wm, 32);
        const v16bf a1 = load_frag_rowK(sA, wm + 16, 32);
        const v16bf b0 = load_frag_rowK(sB, wn, 32);
        const v16bf b1 = load_frag_rowK(sB, wn + 16, 32);
        acc[0][0] = __builtin_amdgcn_wmma_f32_16x16x32_bf16(false, a0, false, b0, (short)0, acc[0][0], false, false);
        acc[0][1] = __builtin_amdgcn_wmma_f32_16x16x32_bf16(false, a0, false, b1, (short)0, acc[0][1], false, false);
        acc[1][0] = __builtin_amdgcn_wmma_f32_16x16x32_bf16(false, a1, false, b0, (short)0, acc[1][0], false, false);
        acc[1][1] = __builtin_amdgcn_wmma_f32_16x16x32_bf16(false, a1, false, b1, (short)0, acc[1][1], false, false);
        __syncthreads();
    }

    // Epilogue: +bias, Q*=scale, scatter to q/k/v head-major bf16.
    const int lane  = threadIdx.x & 31;
    const int nlo   = lane & 15;
    const int madd  = (lane >> 4) * 8;
#pragma unroll
    for (int ti = 0; ti < 2; ++ti) {
#pragma unroll
        for (int tj = 0; tj < 2; ++tj) {
            const int dcol = n0 + wn + tj * 16 + nlo;   // 0..1151
            const int sel  = dcol / CDIM;               // 0:q 1:k 2:v
            const int dd   = dcol - sel * CDIM;
            const int head = dd >> 5;
            const int c    = dd & 31;
            const float b  = bias[dcol];
            unsigned short* dst = (sel == 0) ? q_ws : (sel == 1) ? k_ws : v_ws;
            const float scale = (sel == 0) ? QSCALE : 1.0f;
#pragma unroll
            for (int j = 0; j < 8; ++j) {
                const int m  = m0 + wm + ti * 16 + j + madd;
                const int b_ = m / NTOK;
                const int tk = m - b_ * NTOK;
                const float val = (acc[ti][tj][j] + b) * scale;
                dst[((b_ * NHEADS + head) * NTOK + tk) * HDIM + c] = f32_to_bf16(val);
            }
        }
    }
}

// ---------------------------------------------------------------------------
// Kernel 2: attention. One WG per (window b_, head h): bh = b_*12 + h.
//   S = Q K^T (+mask), softmax, O = P V. S held in LDS fp32, processed in
//   3 blocks of 48 rows to stay under 64KB static LDS. V staged transposed
//   [hd][token] padded to 160 so it serves as the B operand of P@V.
//   Q/K staging is a pure copy -> async-LDS path.
// ---------------------------------------------------------------------------
__global__ __launch_bounds__(256) void attn_kernel(
    const unsigned short* __restrict__ q_ws,
    const unsigned short* __restrict__ k_ws,
    const unsigned short* __restrict__ v_ws,
    const float* __restrict__ mask,
    unsigned short* __restrict__ attn_out) {
    __shared__ __align__(16) unsigned short sQ[NTOK * HDIM];  //  9216 B
    __shared__ __align__(16) unsigned short sK[NTOK * HDIM];  //  9216 B
    __shared__ __align__(16) unsigned short sVt[HDIM * 160];  // 10240 B
    __shared__ __align__(16) float sS[48 * 160];              // 30720 B

    const int tid  = threadIdx.x;
    const int wave = tid >> 5;
    const int bh   = blockIdx.x;
    const int b_   = bh / NHEADS;
    const int h    = bh - b_ * NHEADS;
    const int w    = b_ % NWIN;

    // Stage Q/K (token-major) via async 16B copies; V transposed + zero pad.
    {
        const uint4* qg = (const uint4*)(q_ws + bh * NTOK * HDIM);
        const uint4* kg = (const uint4*)(k_ws + bh * NTOK * HDIM);
        for (int i = tid; i < (NTOK * HDIM) / 8; i += 256) {
            async_copy_b128(&sQ[i * 8], qg + i);
            async_copy_b128(&sK[i * 8], kg + i);
        }
        const unsigned short* vg = v_ws + bh * NTOK * HDIM;
        for (int i = tid; i < HDIM * 160; i += 256) {
            const int c = i / 160;
            const int t = i - c * 160;
            sVt[i] = (t < NTOK) ? vg[t * HDIM + c] : (unsigned short)0;
        }
    }
    async_wait0();
    __syncthreads();

    for (int rb = 0; rb < 3; ++rb) {
        const int r0 = rb * 48;

        // ---- S block: 3 row-tiles x 9 col-tiles, one WMMA each (K=32=hd).
        for (int tile = wave; tile < 27; tile += 8) {
            const int tm = tile / 9, tn = tile - tm * 9;
            const v16bf a = load_frag_rowK(sQ, r0 + tm * 16, HDIM);
            const v16bf b = load_frag_rowK(sK, tn * 16, HDIM);
            v8f c = {};
            c = __builtin_amdgcn_wmma_f32_16x16x32_bf16(false, a, false, b, (short)0, c, false, false);
            const int lane = tid & 31;
            const int col  = tn * 16 + (lane & 15);
            const int rloc = tm * 16 + ((lane >> 4) * 8);
#pragma unroll
            for (int j = 0; j < 8; ++j) sS[(rloc + j) * 160 + col] = c[j];
        }
        __syncthreads();

        // ---- mask + row softmax (one thread per row, fp32).
        if (tid < 48) {
            const int row = r0 + tid;
            const float* mrow = mask + ((size_t)w * NTOK + row) * NTOK;
            float* srow = &sS[tid * 160];
            float mx = -3.0e38f;
            for (int j = 0; j < NTOK; ++j) {
                const float v = srow[j] + mrow[j];
                srow[j] = v;
                mx = fmaxf(mx, v);
            }
            float sum = 0.0f;
            for (int j = 0; j < NTOK; ++j) {
                const float e = __expf(srow[j] - mx);
                srow[j] = e;
                sum += e;
            }
            const float r = 1.0f / sum;
            for (int j = 0; j < NTOK; ++j) srow[j] *= r;
            for (int j = NTOK; j < 160; ++j) srow[j] = 0.0f;   // K-pad for P@V
        }
        __syncthreads();

        // ---- O block: 48x32 = 3x2 tiles; K loop 160 in 5 steps of 32.
        if (wave < 6) {
            const int tm = wave >> 1, tn = (wave & 1) * 16;
            v8f c = {};
#pragma unroll
            for (int kk = 0; kk < 5; ++kk) {
                const v16bf a = load_frag_rowK_f32(sS + kk * 32, tm * 16, 160);
                const v16bf b = load_frag_rowK(sVt + kk * 32, tn, 160);
                c = __builtin_amdgcn_wmma_f32_16x16x32_bf16(false, a, false, b, (short)0, c, false, false);
            }
            const int lane = tid & 31;
            const int col  = tn + (lane & 15);                // 0..31 in head
            const int rloc = tm * 16 + ((lane >> 4) * 8);
#pragma unroll
            for (int j = 0; j < 8; ++j) {
                const int token = r0 + rloc + j;
                attn_out[((size_t)(b_ * NTOK + token)) * CDIM + h * HDIM + col] =
                    f32_to_bf16(c[j]);
            }
        }
        __syncthreads();
    }
}

// ---------------------------------------------------------------------------
// Kernel 3: proj GEMM (M=165888, K=384, N=384) + bias, fp32 output.
// A and B tiles are pure bf16 copies -> async-LDS staging.
// ---------------------------------------------------------------------------
__global__ __launch_bounds__(256) void proj_gemm_kernel(
    const unsigned short* __restrict__ A,    // attn_out bf16 (M, 384)
    const unsigned short* __restrict__ Wb,   // (384, 384) bf16, B[k][n]=Wb[n][k]
    const float* __restrict__ bias,          // (384)
    float* __restrict__ out) {
    __shared__ __align__(16) unsigned short sA[128 * 32];
    __shared__ __align__(16) unsigned short sB[64 * 32];

    const int tid  = threadIdx.x;
    const int wave = tid >> 5;
    const int m0   = blockIdx.y * 128;
    const int n0   = blockIdx.x * 64;
    const int wm   = (wave >> 1) * 32;
    const int wn   = (wave & 1) * 32;

    v8f acc[2][2] = {{{}, {}}, {{}, {}}};

    for (int kt = 0; kt < CDIM; kt += 32) {
        {
            const int row = tid >> 1;
            const int cb  = (tid & 1) * 16;
            const unsigned short* src = A + (size_t)(m0 + row) * CDIM + kt + cb;
            if (kt + 32 < CDIM) __builtin_prefetch(src + 32, 0, 1);
            async_copy_b128(&sA[row * 32 + cb], src);
            async_copy_b128(&sA[row * 32 + cb + 8], src + 8);
        }
        {
            const int n  = tid >> 2;
            const int kb = (tid & 3) * 8;
            async_copy_b128(&sB[n * 32 + kb], Wb + (n0 + n) * CDIM + kt + kb);
        }
        async_wait0();
        __syncthreads();

        const v16bf a0 = load_frag_rowK(sA, wm, 32);
        const v16bf a1 = load_frag_rowK(sA, wm + 16, 32);
        const v16bf b0 = load_frag_rowK(sB, wn, 32);
        const v16bf b1 = load_frag_rowK(sB, wn + 16, 32);
        acc[0][0] = __builtin_amdgcn_wmma_f32_16x16x32_bf16(false, a0, false, b0, (short)0, acc[0][0], false, false);
        acc[0][1] = __builtin_amdgcn_wmma_f32_16x16x32_bf16(false, a0, false, b1, (short)0, acc[0][1], false, false);
        acc[1][0] = __builtin_amdgcn_wmma_f32_16x16x32_bf16(false, a1, false, b0, (short)0, acc[1][0], false, false);
        acc[1][1] = __builtin_amdgcn_wmma_f32_16x16x32_bf16(false, a1, false, b1, (short)0, acc[1][1], false, false);
        __syncthreads();
    }

    const int lane = threadIdx.x & 31;
    const int nlo  = lane & 15;
    const int madd = (lane >> 4) * 8;
#pragma unroll
    for (int ti = 0; ti < 2; ++ti) {
#pragma unroll
        for (int tj = 0; tj < 2; ++tj) {
            const int d = n0 + wn + tj * 16 + nlo;
            const float b = bias[d];
#pragma unroll
            for (int j = 0; j < 8; ++j) {
                const int m = m0 + wm + ti * 16 + j + madd;
                out[(size_t)m * CDIM + d] = acc[ti][tj][j] + b;
            }
        }
    }
}

// ---------------------------------------------------------------------------
// Host launcher. Inputs: x, mask, qkv_w, qkv_b, proj_w, proj_b (all fp32).
// ---------------------------------------------------------------------------
extern "C" void kernel_launch(void* const* d_in, const int* in_sizes, int n_in,
                              void* d_out, int out_size, void* d_ws, size_t ws_size,
                              hipStream_t stream) {
    const float* x      = (const float*)d_in[0];
    const float* mask   = (const float*)d_in[1];
    const float* qkv_w  = (const float*)d_in[2];
    const float* qkv_b  = (const float*)d_in[3];
    const float* proj_w = (const float*)d_in[4];
    const float* proj_b = (const float*)d_in[5];
    float* out = (float*)d_out;

    const size_t headElems = (size_t)BWIN * NHEADS * NTOK * HDIM;  // 63,700,992
    unsigned short* qkvWb  = (unsigned short*)d_ws;
    unsigned short* projWb = qkvWb + (size_t)QKVN * CDIM;
    unsigned short* q_ws   = projWb + (size_t)CDIM * CDIM;
    unsigned short* k_ws   = q_ws + headElems;
    unsigned short* v_ws   = k_ws + headElems;
    unsigned short* attn_o = v_ws + headElems;

    convert_weights_kernel<<<(QKVN * CDIM + 255) / 256, 256, 0, stream>>>(
        qkv_w, proj_w, qkvWb, projWb);

    qkv_gemm_kernel<<<dim3(QKVN / 64, MROWS / 128), 256, 0, stream>>>(
        x, qkvWb, qkv_b, q_ws, k_ws, v_ws);

    attn_kernel<<<BWIN * NHEADS, 256, 0, stream>>>(
        q_ws, k_ws, v_ws, mask, attn_o);

    proj_gemm_kernel<<<dim3(CDIM / 64, MROWS / 128), 256, 0, stream>>>(
        attn_o, projWb, proj_b, out);
}
